// NEATLayer_38448547233842
// MI455X (gfx1250) — matrix operand order repaired and checked by reading, not compiled, not run on hardware
//
#include <hip/hip_runtime.h>

// ---------------- problem constants (match reference) ----------------
#define BATCH   32768
#define NUM_IN  256
#define NUM_HID 512
#define NUM_OUT 64
#define KFAN    32
#define MNODES  (NUM_HID + NUM_OUT)   // 576 non-input nodes
#define NNODES  (NUM_IN + MNODES)     // 832 total nodes

// ---------------- tiling ----------------
#define ROWS    128                   // batch rows per workgroup (8 wave32 x 16 rows)
#define THREADS 256

typedef __attribute__((ext_vector_type(16))) _Float16 v16h;
typedef __attribute__((ext_vector_type(8)))  _Float16 v8h;
typedef __attribute__((ext_vector_type(8)))  float    v8f;
typedef int vint4 __attribute__((vector_size(16)));   // matches builtin param type

typedef vint4 __attribute__((address_space(1)))* gvec4_ptr;  // global int4-vector*
typedef vint4 __attribute__((address_space(3)))* lvec4_ptr;  // LDS int4-vector*

// LDS partition (bytes):
//   state_h : NNODES*ROWS*2       = 212992   (f16 state, [node][row])
//   src_l   : MNODES*KFAN*4       =  73728   (edge_src, async-DMA'd)
//   w_l     : MNODES*KFAN*2       =  36864   (weights converted to f16)
//   nkb_l   : MNODES*4            =   2304   (-log2e * bias, f32)
//   total                         = 325888   (<= 327680 = 320 KB WGP LDS)
#define LDS_STATE_BYTES (NNODES * ROWS * 2)
#define LDS_SRC_BYTES   (MNODES * KFAN * 4)
#define LDS_W_BYTES     (MNODES * KFAN * 2)
#define LDS_NKB_BYTES   (MNODES * 4)
#define LDS_TOTAL       (LDS_STATE_BYTES + LDS_SRC_BYTES + LDS_W_BYTES + LDS_NKB_BYTES)

#define NEG_LOG2E (-1.44269504088896340736f)

__global__ __launch_bounds__(THREADS, 1)
void neat_forward(const float* __restrict__ x,
                  const float* __restrict__ w,
                  const float* __restrict__ bias,
                  const int*   __restrict__ src,
                  float*       __restrict__ out)
{
    extern __shared__ char smem[];
    _Float16* state_h = (_Float16*)smem;
    int*      src_l   = (int*)(smem + LDS_STATE_BYTES);
    _Float16* w_l     = (_Float16*)(smem + LDS_STATE_BYTES + LDS_SRC_BYTES);
    float*    nkb_l   = (float*)(smem + LDS_STATE_BYTES + LDS_SRC_BYTES + LDS_W_BYTES);

    const int tid  = threadIdx.x;
    const int row0 = blockIdx.x * ROWS;

    // ---- stage edge_src: global -> LDS, byte-identical, via async DMA ----
    {
        const int n128 = LDS_SRC_BYTES / 16;   // 4608 b128 transfers
#if __has_builtin(__builtin_amdgcn_global_load_async_to_lds_b128)
        int* src_nc = const_cast<int*>(src);
        for (int i = tid; i < n128; i += THREADS) {
            __builtin_amdgcn_global_load_async_to_lds_b128(
                (gvec4_ptr)(src_nc + i * 4),
                (lvec4_ptr)(src_l + i * 4),
                0, 0);
        }
#else
        for (int i = tid; i < n128; i += THREADS)
            ((int4*)src_l)[i] = ((const int4*)src)[i];
#endif
    }

    // ---- stage weights f32 -> f16 into LDS ----
    {
        const int n4 = (MNODES * KFAN) / 4;    // 4608 float4s
        for (int i = tid; i < n4; i += THREADS) {
            float4 v = ((const float4*)w)[i];
            _Float16* d = w_l + i * 4;
            d[0] = (_Float16)v.x; d[1] = (_Float16)v.y;
            d[2] = (_Float16)v.z; d[3] = (_Float16)v.w;
        }
    }

    // ---- stage biases pre-scaled by -log2e into LDS ----
    {
        const int n4 = MNODES / 4;             // 144 float4s
        for (int i = tid; i < n4; i += THREADS) {
            float4 v = ((const float4*)bias)[i];
            float* d = nkb_l + i * 4;
            d[0] = NEG_LOG2E * v.x; d[1] = NEG_LOG2E * v.y;
            d[2] = NEG_LOG2E * v.z; d[3] = NEG_LOG2E * v.w;
        }
    }

    // ---- stage x transposed f32 -> f16: state_h[col*ROWS + r] ----
    {
        const int r    = tid & (ROWS - 1);
        const int cbeg = (tid >> 7) * (NUM_IN / 2);   // two 128-col segments
        const float* xr = x + (size_t)(row0 + r) * NUM_IN;
        for (int c = cbeg; c < cbeg + NUM_IN / 2; c += 4) {
            float4 v = *(const float4*)(xr + c);      // coalesced-by-thread, L0-hot
            state_h[(c + 0) * ROWS + r] = (_Float16)v.x;
            state_h[(c + 1) * ROWS + r] = (_Float16)v.y;
            state_h[(c + 2) * ROWS + r] = (_Float16)v.z;
            state_h[(c + 3) * ROWS + r] = (_Float16)v.w;
        }
    }

#if __has_builtin(__builtin_amdgcn_s_wait_asynccnt)
    __builtin_amdgcn_s_wait_asynccnt(0);
#else
    asm volatile("s_wait_asynccnt 0" ::: "memory");
#endif
    __syncthreads();

    // ---- sequential DAG evaluation: 16 rows per wave, one WMMA per node ----
    // No inter-wave communication: each wave owns rows [wave*16, wave*16+16),
    // and every gather for row r reads only values previously written for row r.
    const int lane = tid & 31;      // wave32
    const int wave = tid >> 5;      // 0..7
    const int hi   = lane >> 4;     // half-wave select
    const int mrow = lane & 15;     // M (row within 16-tile)
    const int row  = wave * 16 + mrow;

    // software-pipelined bias: load for iteration t+1 alongside the gathers of
    // iteration t so its DS completion is covered by the pre-WMMA dscnt wait.
    float nkb = nkb_l[0];

    for (int t = 0; t < MNODES; ++t) {
        const int tn = (t + 1 < MNODES) ? (t + 1) : t;
        float nkb_next = nkb_l[tn];

        // A-matrix k-set for this lane: [8*hi, 8*hi+8) and [16+8*hi, 16+8*hi+8)
        const int* sp = src_l + t * KFAN + 8 * hi;
        int4 sA = *(const int4*)(sp);
        int4 sB = *(const int4*)(sp + 4);
        int4 sC = *(const int4*)(sp + 16);
        int4 sD = *(const int4*)(sp + 20);
        int s[16] = { sA.x, sA.y, sA.z, sA.w, sB.x, sB.y, sB.z, sB.w,
                      sC.x, sC.y, sC.z, sC.w, sD.x, sD.y, sD.z, sD.w };

        v16h a;
#pragma unroll
        for (int h = 0; h < 16; ++h)
            a[h] = state_h[s[h] * ROWS + row];          // f16 LDS gather

        // B-matrix: same k-set as A; replicate across all 16 columns
        const _Float16* wk = w_l + t * KFAN + 8 * hi;
        v16h bm;
#pragma unroll
        for (int h = 0; h < 16; ++h) bm[h] = wk[h + (h & 8)];

        // C = 0 -> inline-0 SRC2 encoding; bias folded into sigmoid below.
        v8f c = {};
        c = __builtin_amdgcn_wmma_f32_16x16x32_f16(false, a, false, bm,
                                                   (short)0, c, false, false);

        // D layout: VGPR g holds row (g + 8*hi), replicated over columns.
        // Lanes 0 and 16 each own 8 rows -> sigmoid + one ds_store_b128.
        // sigmoid(d + b) = 1 / (1 + exp2(fma(-log2e, d, -log2e*b)))
        if (mrow == 0) {
            v8h stv;
#pragma unroll
            for (int g = 0; g < 8; ++g) {
                float e  = __builtin_amdgcn_exp2f(__builtin_fmaf(NEG_LOG2E, c[g], nkb));
                float sg = __builtin_amdgcn_rcpf(1.0f + e);
                stv[g] = (_Float16)sg;
            }
            *(v8h*)(state_h + (NUM_IN + t) * ROWS + wave * 16 + 8 * hi) = stv;
        }
        nkb = nkb_next;
    }
    __syncthreads();

    // ---- epilogue: output nodes (cols 768..831) -> d_out as f32 ----
    {
        const int r    = tid & (ROWS - 1);
        const int cbeg = (tid >> 7) * (NUM_OUT / 2);   // 32 cols per thread
        float* orow = out + (size_t)(row0 + r) * NUM_OUT;
        for (int c = cbeg; c < cbeg + NUM_OUT / 2; ++c)
            orow[c] = (float)state_h[(NUM_IN + NUM_HID + c) * ROWS + r];
    }
}

extern "C" void kernel_launch(void* const* d_in, const int* in_sizes, int n_in,
                              void* d_out, int out_size, void* d_ws, size_t ws_size,
                              hipStream_t stream)
{
    (void)in_sizes; (void)n_in; (void)out_size; (void)d_ws; (void)ws_size;
    const float* x  = (const float*)d_in[0];   // [B, 256]
    const float* w  = (const float*)d_in[1];   // [576, 32]
    const float* b  = (const float*)d_in[2];   // [576]
    const int*   es = (const int*)d_in[3];     // [576, 32]
    float* out = (float*)d_out;                // [B, 64]

    // 325.9 KB dynamic LDS can exceed the default dynamic-shared cap; opt in.
    static_assert(LDS_TOTAL <= 327680, "exceeds 320 KB WGP LDS");
    (void)hipFuncSetAttribute((const void*)neat_forward,
                              hipFuncAttributeMaxDynamicSharedMemorySize,
                              LDS_TOTAL);

    dim3 grid(BATCH / ROWS);                   // 256 workgroups, one per WGP pass
    neat_forward<<<grid, THREADS, LDS_TOTAL, stream>>>(x, w, b, es, out);
}
